// NIFCU_57552561766454
// MI455X (gfx1250) — compile-verified, matrix operand order checked by reading, not run on hardware
//
#include <hip/hip_runtime.h>
#include <hip/hip_bf16.h>
#include <stdint.h>

#define B_  8
#define S_  2048
#define E_  2048
#define H_  16
#define DH_ 128
#define M_  1024

typedef __attribute__((ext_vector_type(16))) __bf16 v16bf;
typedef __attribute__((ext_vector_type(8)))  __bf16 bf16x8;
typedef __attribute__((ext_vector_type(8)))  float  v8f;

#if defined(__gfx1250__) && __has_builtin(__builtin_amdgcn_s_wait_asynccnt)
#define HAVE_ASYNC 1
#else
#define HAVE_ASYNC 0
#endif

__device__ __forceinline__ __bf16 f2bf(float f) {
  union { float f; uint32_t u; } v; v.f = f;
  uint32_t r = v.u + 0x7FFFu + ((v.u >> 16) & 1u);   // round-to-nearest-even
  uint16_t h = (uint16_t)(r >> 16);
  return __builtin_bit_cast(__bf16, h);
}

// 16B global -> LDS copy, ASYNCcnt-tracked (CDNA5 GLOBAL_LOAD_ASYNC_TO_LDS).
// VDST = LDS byte offset (low 32 bits of generic LDS pointer; HW adds LDS_BASE),
// VADDR = 64-bit global address in a VGPR pair.
__device__ __forceinline__ void cp_b128(const __bf16* g, __bf16* l) {
#if HAVE_ASYNC
  unsigned l32 = (unsigned)(uintptr_t)l;
  asm volatile("global_load_async_to_lds_b128 %0, %1, off"
               :
               : "v"(l32), "v"(g)
               : "memory");
#else
  *(bf16x8*)l = *(const bf16x8*)g;
#endif
}

template <int N>
__device__ __forceinline__ void wait_async() {
#if HAVE_ASYNC
  __builtin_amdgcn_s_wait_asynccnt(N);
#endif
}

// Build a 16x(K=32) bf16 fragment from two contiguous 8-element (16B) chunks.
// Per ISA layout: element i <-> K = (i&7) + (lane>=16)*8 + (i>=8)*16.
__device__ __forceinline__ v16bf make_frag(const __bf16* p) {
  bf16x8 lo = *(const bf16x8*)p;
  bf16x8 hi = *(const bf16x8*)(p + 16);
  v16bf r;
#pragma unroll
  for (int i = 0; i < 8; ++i) { r[i] = lo[i]; r[i + 8] = hi[i]; }
  return r;
}

__device__ __forceinline__ v8f wmma_bf16(v16bf a, v16bf b, v8f c) {
  return __builtin_amdgcn_wmma_f32_16x16x32_bf16(false, a, false, b, (short)0, c,
                                                 false, false);
}

// ---------------- elementwise conversion kernels ----------------
__global__ __launch_bounds__(256)
void cvt_f32_bf16(const float* __restrict__ in, __bf16* __restrict__ out, int n) {
  for (int i = blockIdx.x * 256 + threadIdx.x; i < n; i += gridDim.x * 256)
    out[i] = f2bf(in[i]);
}

__global__ __launch_bounds__(256)
void addcvt_f32_bf16(const float* __restrict__ a, const float* __restrict__ b,
                     __bf16* __restrict__ out, int n) {
  for (int i = blockIdx.x * 256 + threadIdx.x; i < n; i += gridDim.x * 256)
    out[i] = f2bf(a[i] + b[i]);
}

// Tiled transpose + convert: in (batch, K, N) f32  ->  out (batch, N, K) bf16.
// One-time cost so the GEMM/attention hot loops only do straight async copies.
__global__ __launch_bounds__(256)
void cvtT_f32_bf16(const float* __restrict__ in, __bf16* __restrict__ out,
                   int K, int N) {
  __shared__ float t[32][33];
  const int nb = N / 32;
  const int n0 = (blockIdx.x % nb) * 32;
  const int k0 = (blockIdx.x / nb) * 32;
  const float* inB = in + (size_t)blockIdx.y * K * N;
  __bf16* outB = out + (size_t)blockIdx.y * N * K;
  const int lx = threadIdx.x & 31, ly = threadIdx.x >> 5;
#pragma unroll
  for (int r = 0; r < 32; r += 8)
    t[r + ly][lx] = inB[(size_t)(k0 + r + ly) * N + n0 + lx];
  __syncthreads();
#pragma unroll
  for (int r = 0; r < 32; r += 8)
    outB[(size_t)(n0 + r + ly) * K + k0 + lx] = f2bf(t[lx][r + ly]);
}

// ---------------- 128x128 bf16 GEMM, f32 accumulate, async double-buffer ----
// C[M,N] = A[M,K] * BT[N,K]^T + bias[N] ; OUT_BF16 selects bf16 vs f32 output.
template <int OUT_BF16>
__global__ __launch_bounds__(256)
void gemm128(const __bf16* __restrict__ A, const __bf16* __restrict__ BT,
             const float* __restrict__ bias, float* __restrict__ Cf,
             __bf16* __restrict__ Cb, int Mrows, int N, int K) {
  __shared__ __bf16 as_[2][128 * 32];   // [m][k]
  __shared__ __bf16 bs_[2][128 * 32];   // [n][k]

  const int tid = threadIdx.x;
  const int wid = tid >> 5, lane = tid & 31;
  const int hlf = lane >> 4, ln = lane & 15;
  const int m0 = blockIdx.y * 128, n0 = blockIdx.x * 128;
  const int wr = (wid >> 1) * 32;    // 4 wave-rows of 32
  const int wc = (wid & 1) * 64;     // 2 wave-cols of 64
  const int nk = K / 32;

  // this thread's tile-copy coordinates (2 x 16B for A, 2 x 16B for B)
  const int r0 = tid >> 2, kc0 = (tid & 3) * 8;          // chunks 0..255
  const int r1 = (tid + 256) >> 2, kc1 = ((tid + 256) & 3) * 8;  // 256..511

  auto issue = [&](int k0c, int buf) {
    cp_b128(&A[(size_t)(m0 + r0) * K + k0c + kc0], &as_[buf][r0 * 32 + kc0]);
    cp_b128(&A[(size_t)(m0 + r1) * K + k0c + kc1], &as_[buf][r1 * 32 + kc1]);
    cp_b128(&BT[(size_t)(n0 + r0) * K + k0c + kc0], &bs_[buf][r0 * 32 + kc0]);
    cp_b128(&BT[(size_t)(n0 + r1) * K + k0c + kc1], &bs_[buf][r1 * 32 + kc1]);
  };

  v8f acc[8] = {};

  issue(0, 0);                                   // prologue: chunk 0 -> buf 0
  for (int c = 0; c < nk; ++c) {
    const int buf = c & 1;
    wait_async<0>();
    __syncthreads();                             // tile c resident everywhere
    if (c + 1 < nk) issue((c + 1) * 32, buf ^ 1);  // overlap next copy w/ WMMA

    v16bf af[2], bf[4];
#pragma unroll
    for (int rt = 0; rt < 2; ++rt)
      af[rt] = make_frag(&as_[buf][(wr + rt * 16 + ln) * 32 + hlf * 8]);
#pragma unroll
    for (int ct = 0; ct < 4; ++ct)
      bf[ct] = make_frag(&bs_[buf][(wc + ct * 16 + ln) * 32 + hlf * 8]);
#pragma unroll
    for (int rt = 0; rt < 2; ++rt)
#pragma unroll
      for (int ct = 0; ct < 4; ++ct)
        acc[rt * 4 + ct] = wmma_bf16(af[rt], bf[ct], acc[rt * 4 + ct]);
  }

  // epilogue: C layout = VGPR e -> row e + hlf*8 ; lane&15 -> col
#pragma unroll
  for (int rt = 0; rt < 2; ++rt)
#pragma unroll
    for (int ct = 0; ct < 4; ++ct) {
      int col = n0 + wc + ct * 16 + ln;
      float bv = bias[col];
#pragma unroll
      for (int e = 0; e < 8; ++e) {
        int row = m0 + wr + rt * 16 + e + hlf * 8;
        float v = acc[rt * 4 + ct][e] + bv;
        if (OUT_BF16) Cb[(size_t)row * N + col] = f2bf(v);
        else          Cf[(size_t)row * N + col] = v;
      }
    }
}

// ---------------- fused attention: scores + online softmax + AV ----------------
// q: (B,S,H*Dh) bf16 ; kb: (H,M,Dh) bf16 ; vtb: (H,Dh,M) bf16 (pre-transposed)
// ao: (B,S,H*Dh) bf16
__global__ __launch_bounds__(256)
void attn_kernel(const __bf16* __restrict__ q, const __bf16* __restrict__ kb,
                 const __bf16* __restrict__ vtb, __bf16* __restrict__ ao) {
  const int sTiles = S_ / 128;
  const int st = blockIdx.x % sTiles;
  const int h  = (blockIdx.x / sTiles) % H_;
  const int b  = blockIdx.x / (sTiles * H_);

  const int tid = threadIdx.x;
  const int w = tid >> 5, lane = tid & 31;
  const int hlf = lane >> 4, ln = lane & 15;
  const int s0 = st * 128 + w * 16;      // this wave's 16 rows

  __shared__ __bf16 ks_[64 * 128];       // [m_local][d]   (B-frag for scores)
  __shared__ __bf16 vt_[128 * 64];       // [d][m_local]   (B-frag for AV)
  __shared__ __bf16 ps_[8][16 * 64];     // per-wave probs [s_in_wave][m_local]

  // q fragments for this wave's 16 rows, held in registers (4 K-steps over Dh)
  v16bf qf[4];
  const __bf16* qrow = q + ((size_t)(b * S_ + s0 + ln) * (size_t)E_) + h * DH_;
#pragma unroll
  for (int kk = 0; kk < 4; ++kk)
    qf[kk] = make_frag(qrow + kk * 32 + hlf * 8);

  float runmax[8], runsum[8];
  v8f oacc[8] = {};
#pragma unroll
  for (int r = 0; r < 8; ++r) { runmax[r] = -3.0e38f; runsum[r] = 0.0f; }

  const float sscale = 0.022097086912079608f;   // 1/sqrt(E=2048)
  const __bf16* kbase  = kb  + (size_t)h * M_ * DH_;
  const __bf16* vtbase = vtb + (size_t)h * DH_ * M_;

  for (int mc = 0; mc < M_; mc += 64) {
    __syncthreads();                       // buffers free (prev chunk consumed)
    // issue K-chunk copies (4 per thread), then V-chunk copies (4 per thread)
#pragma unroll
    for (int i = tid; i < 1024; i += 256) {
      int r = i >> 4, dc = (i & 15) * 8;   // [m][d] direct copy
      cp_b128(&kbase[(size_t)(mc + r) * DH_ + dc], &ks_[r * 128 + dc]);
    }
#pragma unroll
    for (int i = tid; i < 1024; i += 256) {
      int d = i >> 3, m8 = (i & 7) * 8;    // [d][m] direct copy (pre-transposed)
      cp_b128(&vtbase[(size_t)d * M_ + mc + m8], &vt_[d * 64 + m8]);
    }
    wait_async<4>();                       // K tile landed; V still in flight
    __syncthreads();

    // scores: 16 rows x 64 m-cols, 4 n-tiles x 4 K-steps over Dh
    v8f sacc[4] = {};
#pragma unroll
    for (int j = 0; j < 4; ++j)
#pragma unroll
      for (int kk = 0; kk < 4; ++kk) {
        v16bf bf = make_frag(&ks_[(j * 16 + ln) * 128 + kk * 32 + hlf * 8]);
        sacc[j] = wmma_bf16(qf[kk], bf, sacc[j]);
      }

    // online softmax per row (row r + hlf*8 spans lanes of this 16-lane half)
#pragma unroll
    for (int r = 0; r < 8; ++r) {
      float sv[4], pv[4];
      float mx = -3.0e38f;
#pragma unroll
      for (int j = 0; j < 4; ++j) { sv[j] = sacc[j][r] * sscale; mx = fmaxf(mx, sv[j]); }
#pragma unroll
      for (int off = 1; off < 16; off <<= 1) mx = fmaxf(mx, __shfl_xor(mx, off));
      float nm = fmaxf(runmax[r], mx);
      float corr = __expf(runmax[r] - nm);
      runmax[r] = nm;
      float rs = 0.0f;
#pragma unroll
      for (int j = 0; j < 4; ++j) { pv[j] = __expf(sv[j] - nm); rs += pv[j]; }
#pragma unroll
      for (int off = 1; off < 16; off <<= 1) rs += __shfl_xor(rs, off);
      runsum[r] = runsum[r] * corr + rs;
#pragma unroll
      for (int t = 0; t < 8; ++t) oacc[t][r] *= corr;
      int srow = r + hlf * 8;
#pragma unroll
      for (int j = 0; j < 4; ++j) ps_[w][srow * 64 + j * 16 + ln] = f2bf(pv[j]);
    }

    wait_async<0>();                       // V tile landed (overlapped above)
    __syncthreads();

    // AV: A = probs (16 x 64), B = V^T tiles ; 2 K-steps over m, 8 d-tiles
#pragma unroll
    for (int kk = 0; kk < 2; ++kk) {
      v16bf af = make_frag(&ps_[w][ln * 64 + kk * 32 + hlf * 8]);
#pragma unroll
      for (int t = 0; t < 8; ++t) {
        v16bf bf = make_frag(&vt_[(t * 16 + ln) * 64 + kk * 32 + hlf * 8]);
        oacc[t] = wmma_bf16(af, bf, oacc[t]);
      }
    }
  }

  // normalize and write (B,S,H*Dh) bf16 for the dehead GEMM
#pragma unroll
  for (int t = 0; t < 8; ++t)
#pragma unroll
    for (int e = 0; e < 8; ++e) {
      int s = s0 + e + hlf * 8;
      int d = t * 16 + ln;
      float v = oacc[t][e] * (1.0f / runsum[e]);
      ao[(size_t)(b * S_ + s) * (size_t)E_ + h * DH_ + d] = f2bf(v);
    }
}

// ---------------- host launch ----------------
extern "C" void kernel_launch(void* const* d_in, const int* in_sizes, int n_in,
                              void* d_out, int out_size, void* d_ws, size_t ws_size,
                              hipStream_t stream) {
  (void)in_sizes; (void)n_in; (void)out_size; (void)ws_size;
  const float* tensor = (const float*)d_in[0];
  const float* Wq     = (const float*)d_in[1];
  const float* bq     = (const float*)d_in[2];
  const float* keyp   = (const float*)d_in[3];
  const float* calib  = (const float*)d_in[4];
  const float* value  = (const float*)d_in[5];
  const float* Wd     = (const float*)d_in[6];
  const float* bd     = (const float*)d_in[7];
  float* out = (float*)d_out;

  char* ws = (char*)d_ws;
  size_t off = 0;
  auto alloc = [&](size_t bytes) -> void* {
    void* p = ws + off;
    off += (bytes + 255) & ~(size_t)255;
    return p;
  };
  __bf16* tA  = (__bf16*)alloc((size_t)B_ * S_ * E_ * 2);   // tensor bf16
  __bf16* qb  = (__bf16*)alloc((size_t)B_ * S_ * E_ * 2);   // q (B,S,H*Dh)
  __bf16* ao  = (__bf16*)alloc((size_t)B_ * S_ * E_ * 2);   // attn out
  __bf16* wqT = (__bf16*)alloc((size_t)E_ * H_ * DH_ * 2);  // Wq^T (N x K)
  __bf16* wdT = (__bf16*)alloc((size_t)H_ * DH_ * E_ * 2);  // Wd^T (N x K)
  __bf16* kbf = (__bf16*)alloc((size_t)H_ * M_ * DH_ * 2);  // key+calib (H,M,Dh)
  __bf16* vtb = (__bf16*)alloc((size_t)H_ * M_ * DH_ * 2);  // V^T (H,Dh,M)

  cvt_f32_bf16<<<2048, 256, 0, stream>>>(tensor, tA, B_ * S_ * E_);
  // Wq: (K=E) x (N=H*Dh) -> N x K
  cvtT_f32_bf16<<<dim3((2048 / 32) * (E_ / 32), 1), 256, 0, stream>>>(Wq, wqT, E_, 2048);
  // Wd: (K=H*Dh) x (N=E) -> N x K
  cvtT_f32_bf16<<<dim3((E_ / 32) * (2048 / 32), 1), 256, 0, stream>>>(Wd, wdT, 2048, E_);
  addcvt_f32_bf16<<<512, 256, 0, stream>>>(keyp, calib, kbf, H_ * M_ * DH_);
  // V: per head (K=M) x (N=Dh) -> Dh x M
  cvtT_f32_bf16<<<dim3((DH_ / 32) * (M_ / 32), H_), 256, 0, stream>>>(value, vtb, M_, DH_);

  // q = tensor * Wq + bq   (bf16 out)
  gemm128<1><<<dim3(E_ / 128, (B_ * S_) / 128), 256, 0, stream>>>(
      tA, wqT, bq, nullptr, qb, B_ * S_, E_, E_);
  // fused attention
  attn_kernel<<<dim3(B_ * H_ * (S_ / 128)), 256, 0, stream>>>(qb, kbf, vtb, ao);
  // out = ao * Wd + bd     (f32 out)
  gemm128<0><<<dim3(E_ / 128, (B_ * S_) / 128), 256, 0, stream>>>(
      ao, wdT, bd, out, nullptr, B_ * S_, E_, E_);
}
